// FourierQLSTMCell_2104533975410
// MI455X (gfx1250) — matrix operand order repaired and checked by reading, not compiled, not run on hardware
//
#include <hip/hip_runtime.h>
#include <hip/hip_bf16.h>

// ---------------- problem constants ----------------
#define NQ       10
#define DIM      1024          // 2^NQ
#define HID      10
#define DEPTH_L  3
#define WIN      64
#define NFREQ    33
#define FFTD     66
#define BATCH_N  1024
#define PI_F     3.14159265358979323846f

typedef __attribute__((ext_vector_type(16))) _Float16 v16h;
typedef __attribute__((ext_vector_type(8)))  float    v8f;

// =====================================================================
// Kernel 1: DFT features (via WMMA f16->f32 GEMM) + input/hidden
// projections -> comb[b][q].  One wave (32 threads) per 16 batch rows.
//
//   Re[m][k] = sum_n x[m][n] *  cos(2*pi*k*n/64)
//   Im[m][k] = sum_n x[m][n] * -sin(2*pi*k*n/64)      (rfft convention)
//
// GEMM: A = x tile (16x64, f16), B = twiddle tables (64x48, f16, 3
// column tiles of 16 freqs; freqs 33..47 are padding), C/D = f32.
// =====================================================================
__global__ __launch_bounds__(32) void k_features(
    const float* __restrict__ x,        // [1024][64]
    const float* __restrict__ c_mag,    // [1024][10]
    const float* __restrict__ c_phase,  // [1024][10]
    const float* __restrict__ Wi,       // [10][66]
    const float* __restrict__ bi,       // [10]
    const float* __restrict__ Wh,       // [10][20]
    const float* __restrict__ bh,       // [10]
    float* __restrict__ comb)           // [1024][16] (10 used)
{
  __shared__ _Float16 cosT[WIN][48];
  __shared__ _Float16 sinT[WIN][48];    // holds -sin
  __shared__ float    reF[16][48];
  __shared__ float    imF[16][48];
  __shared__ float    feat[16][FFTD];

  const int lane = threadIdx.x;         // 0..31 (wave32)
  const int row0 = blockIdx.x * 16;     // 64 blocks

  // hint-prefetch the x tile (emits global_prefetch_b8)
  __builtin_prefetch(x + (row0 + (lane & 15)) * WIN, 0, 1);

  // ---- build twiddle tables in LDS ----
  for (int idx = lane; idx < WIN * 48; idx += 32) {
    int n = idx / 48;                   // time sample
    int k = idx % 48;                   // frequency
    float ang = 2.0f * PI_F * (float)(n * k) / (float)WIN;
    float s, c;
    __sincosf(ang, &s, &c);
    cosT[n][k] = (_Float16)c;
    sinT[n][k] = (_Float16)(-s);
  }
  __syncthreads();

  // ---- pack A fragments (16-bit A 16x32 layout, ISA 7.12.2) ----
  // lane L: M = L&15 ; lanes 16-31 hold the K+8 / K+24 halves.
  v16h afr[2];
  {
    const int  M  = lane & 15;
    const int  hb = (lane >> 4) & 1;    // half-select
    const float* xr = x + (row0 + M) * WIN;
    for (int ks = 0; ks < 2; ++ks) {
      for (int j = 0; j < 16; ++j) {
        int v = j >> 1, h = j & 1;
        int kl = (v < 4) ? (hb * 8 + 2 * v + h)
                         : (16 + hb * 8 + 2 * (v - 4) + h);
        afr[ks][j] = (_Float16)xr[ks * 32 + kl];
      }
    }
  }

  // ---- 3 column tiles x {cos,sin} x 2 k-steps = 12 WMMAs ----
  {
    const int nloc  = lane & 15;              // output column within tile
    const int halfK = (lane >> 4) ? 8 : 0;    // B: lanes 16-31 hold K+8 / K+24 halves
    for (int ct = 0; ct < 3; ++ct) {
      v8f accR = {}; v8f accI = {};
      for (int ks = 0; ks < 2; ++ks) {
        v16h bR, bI;
        for (int j = 0; j < 16; ++j) {
          int v = j >> 1, h = j & 1;
          int kg = ks * 32 + ((v < 4) ? (halfK + 2 * v + h) : (16 + halfK + 2 * (v - 4) + h)); // time index n
          int fg = ct * 16 + nloc;              // freq index k
          bR[j] = cosT[kg][fg];
          bI[j] = sinT[kg][fg];
        }
        accR = __builtin_amdgcn_wmma_f32_16x16x32_f16(
                 false, afr[ks], false, bR, (short)0, accR, false, false);
        accI = __builtin_amdgcn_wmma_f32_16x16x32_f16(
                 false, afr[ks], false, bI, (short)0, accI, false, false);
      }
      // f32 C/D layout: VGPR r holds row M = r + 8*(lane/16), col = lane&15
      for (int r = 0; r < 8; ++r) {
        int m = r + ((lane >> 4) ? 8 : 0);
        int f = ct * 16 + nloc;
        reF[m][f] = accR[r];
        imF[m][f] = accI[r];
      }
    }
  }
  __syncthreads();

  // ---- magnitude / phase features ----
  for (int idx = lane; idx < 16 * NFREQ; idx += 32) {
    int m = idx / NFREQ, k = idx % NFREQ;
    float re = reF[m][k], im = imF[m][k];
    feat[m][k]         = log1pf(sqrtf(re * re + im * im));
    feat[m][NFREQ + k] = atan2f(im, re) * (1.0f / PI_F);
  }
  __syncthreads();

  // ---- x_proj + h_proj -> comb ----
  for (int idx = lane; idx < 16 * NQ; idx += 32) {
    int m = idx / NQ, q = idx % NQ;
    int b = row0 + m;
    float acc = bi[q];
    for (int j = 0; j < FFTD; ++j) acc += feat[m][j] * Wi[q * FFTD + j];
    float xp = tanhf(acc) * PI_F;
    float acch = bh[q];
    for (int j = 0; j < HID; ++j) acch += c_mag[b * HID + j]   * Wh[q * 20 + j];
    for (int j = 0; j < HID; ++j) acch += c_phase[b * HID + j] * Wh[q * 20 + HID + j];
    float hp = tanhf(acch) * PI_F;
    comb[b * 16 + q] = xp + hp;
  }
}

// =====================================================================
// Kernel 2: 10-qubit statevector simulation. One block = one
// (batch, gate) pair; grid = (1024, 4). State (1024 complex) in LDS.
//  - per qubit: fused U = RX(fs*th) * RY(th) * H   (one pass)
//  - per layer: CNOT chain == Gray-code gather new[j]=old[j^(j>>1)],
//    then RY(vp[l][i]) per qubit.
//  - Z expectations reduced deterministically via LDS partials, serially summed per qubit.
// =====================================================================
__global__ __launch_bounds__(128) void k_vqc(
    const float* __restrict__ comb,   // [1024][16]
    const float* __restrict__ fs_i, const float* __restrict__ vp_i,
    const float* __restrict__ fs_f, const float* __restrict__ vp_f,
    const float* __restrict__ fs_g, const float* __restrict__ vp_g,
    const float* __restrict__ fs_o, const float* __restrict__ vp_o,
    float* __restrict__ expect)       // [4][1024][16] (10 used)
{
  __shared__ float sre[DIM];
  __shared__ float sim_[DIM];
  __shared__ float partial[NQ][128];

  const int b = blockIdx.x;
  const int g = blockIdx.y;
  const int t = threadIdx.x;

  const float* fs_tab[4] = { fs_i, fs_f, fs_g, fs_o };
  const float* vp_tab[4] = { vp_i, vp_f, vp_g, vp_o };
  const float* fs = fs_tab[g];
  const float* vp = vp_tab[g];

  // |0...0>
  for (int j = t; j < DIM; j += 128) { sre[j] = 0.0f; sim_[j] = 0.0f; }
  if (t == 0) sre[0] = 1.0f;
  __syncthreads();

  const float hsc = 0.70710678118654752440f;

  // ---- encoding: U = RX * RY * H per qubit ----
  for (int i = 0; i < NQ; ++i) {
    float th = comb[b * 16 + i];
    float ph = fs[i] * th;
    float c1, s1, c2, s2;
    __sincosf(th * 0.5f, &s1, &c1);
    __sincosf(ph * 0.5f, &s2, &c2);
    float A_ = hsc * (c1 - s1), B_ = hsc * (c1 + s1);
    // U00=(c2*A, -s2*B)  U01=(c2*B, s2*A)  U10=(c2*B, -s2*A)  U11=(-c2*A, -s2*B)
    float u00r =  c2 * A_, u00i = -s2 * B_;
    float u01r =  c2 * B_, u01i =  s2 * A_;
    float u10r =  c2 * B_, u10i = -s2 * A_;
    float u11r = -c2 * A_, u11i = -s2 * B_;

    const int p = 9 - i;                 // qubit i toggles bit (9-i)
    const unsigned mask = 1u << p;
    for (int tt = t; tt < DIM / 2; tt += 128) {
      unsigned lo = ((unsigned)(tt >> p) << (p + 1)) | ((unsigned)tt & (mask - 1u));
      unsigned hi = lo | mask;
      float r0 = sre[lo], i0 = sim_[lo];
      float r1 = sre[hi], i1 = sim_[hi];
      sre[lo] = u00r * r0 - u00i * i0 + u01r * r1 - u01i * i1;
      sim_[lo] = u00r * i0 + u00i * r0 + u01r * i1 + u01i * r1;
      sre[hi] = u10r * r0 - u10i * i0 + u11r * r1 - u11i * i1;
      sim_[hi] = u10r * i0 + u10i * r0 + u11r * i1 + u11i * r1;
    }
    __syncthreads();
  }

  // ---- variational layers ----
  for (int l = 0; l < DEPTH_L; ++l) {
    // CNOT chain as one permutation pass
    float tr[DIM / 128], ti[DIM / 128];
    for (int k = 0; k < DIM / 128; ++k) {
      int j = t + k * 128;
      int src = j ^ (j >> 1);
      tr[k] = sre[src]; ti[k] = sim_[src];
    }
    __syncthreads();
    for (int k = 0; k < DIM / 128; ++k) {
      int j = t + k * 128;
      sre[j] = tr[k]; sim_[j] = ti[k];
    }
    __syncthreads();

    // RY rotations
    for (int i = 0; i < NQ; ++i) {
      float th = vp[l * NQ + i];
      float c, s;
      __sincosf(th * 0.5f, &s, &c);
      const int p = 9 - i;
      const unsigned mask = 1u << p;
      for (int tt = t; tt < DIM / 2; tt += 128) {
        unsigned lo = ((unsigned)(tt >> p) << (p + 1)) | ((unsigned)tt & (mask - 1u));
        unsigned hi = lo | mask;
        float r0 = sre[lo], i0 = sim_[lo];
        float r1 = sre[hi], i1 = sim_[hi];
        sre[lo] = c * r0 - s * r1;  sim_[lo] = c * i0 - s * i1;
        sre[hi] = s * r0 + c * r1;  sim_[hi] = s * i0 + c * i1;
      }
      __syncthreads();
    }
  }

  // ---- Z expectations (deterministic reduction) ----
  float part[NQ];
  for (int i = 0; i < NQ; ++i) part[i] = 0.0f;
  for (int j = t; j < DIM; j += 128) {
    float pr = sre[j] * sre[j] + sim_[j] * sim_[j];
    for (int i = 0; i < NQ; ++i)
      part[i] += ((j >> (9 - i)) & 1) ? -pr : pr;
  }
  for (int i = 0; i < NQ; ++i) partial[i][t] = part[i];
  __syncthreads();
  if (t < NQ) {
    float s = 0.0f;
    for (int k = 0; k < 128; ++k) s += partial[t][k];
    expect[(g * BATCH_N + b) * 16 + t] = s;
  }
}

// =====================================================================
// Kernel 3: LSTM-style combine + output projection.
// out layout: [0,1024) out ; [1024,11264) c_mag_new ; [11264,21504) c_phase_new
// =====================================================================
__global__ __launch_bounds__(256) void k_cell(
    const float* __restrict__ expect,   // [4][1024][16]
    const float* __restrict__ c_mag,    // [1024][10]
    const float* __restrict__ c_phase,  // [1024][10]
    const float* __restrict__ Wo,       // [1][10]
    const float* __restrict__ bo,       // [1]
    float* __restrict__ out)
{
  int b = blockIdx.x * blockDim.x + threadIdx.x;
  if (b >= BATCH_N) return;
  float acc = bo[0];
  for (int h = 0; h < HID; ++h) {
    float ei = expect[(0 * BATCH_N + b) * 16 + h];
    float ef = expect[(1 * BATCH_N + b) * 16 + h];
    float eg = expect[(2 * BATCH_N + b) * 16 + h];
    float eo = expect[(3 * BATCH_N + b) * 16 + h];
    float it = (ei + 1.0f) * 0.5f;
    float ft = (ef + 1.0f) * 0.5f;
    float gt = eg;
    float ot = (eo + 1.0f) * 0.5f;
    float cm = c_mag[b * HID + h];
    float cp = c_phase[b * HID + h];
    float cmn = ft * cm + it * fabsf(gt);
    float sg, cg;
    __sincosf(gt * PI_F, &sg, &cg);
    float delta = atan2f(sg, cg) * (1.0f / PI_F);   // wrap gt into (-1,1]
    float cpn = cp + it * delta;
    cpn = fmodf(cpn + 1.0f, 2.0f);
    if (cpn < 0.0f) cpn += 2.0f;
    cpn -= 1.0f;
    float ht = ot * cmn * __cosf(cpn * PI_F);
    acc += ht * Wo[h];
    out[BATCH_N + b * HID + h]                    = cmn;
    out[BATCH_N + BATCH_N * HID + b * HID + h]    = cpn;
  }
  out[b] = acc;
}

// =====================================================================
extern "C" void kernel_launch(void* const* d_in, const int* in_sizes, int n_in,
                              void* d_out, int out_size, void* d_ws, size_t ws_size,
                              hipStream_t stream) {
  const float* x       = (const float*)d_in[0];
  const float* c_mag   = (const float*)d_in[1];
  const float* c_phase = (const float*)d_in[2];
  const float* Wi      = (const float*)d_in[3];
  const float* bi      = (const float*)d_in[4];
  const float* Wh      = (const float*)d_in[5];
  const float* bh      = (const float*)d_in[6];
  const float* fs_i    = (const float*)d_in[7];
  const float* vp_i    = (const float*)d_in[8];
  const float* fs_f    = (const float*)d_in[9];
  const float* vp_f    = (const float*)d_in[10];
  const float* fs_g    = (const float*)d_in[11];
  const float* vp_g    = (const float*)d_in[12];
  const float* fs_o    = (const float*)d_in[13];
  const float* vp_o    = (const float*)d_in[14];
  const float* Wo      = (const float*)d_in[15];
  const float* bo      = (const float*)d_in[16];

  float* out    = (float*)d_out;
  float* comb   = (float*)d_ws;             // 1024*16 f32
  float* expect = comb + BATCH_N * 16;      // 4*1024*16 f32

  k_features<<<dim3(BATCH_N / 16), dim3(32), 0, stream>>>(
      x, c_mag, c_phase, Wi, bi, Wh, bh, comb);
  k_vqc<<<dim3(BATCH_N, 4), dim3(128), 0, stream>>>(
      comb, fs_i, vp_i, fs_f, vp_f, fs_g, vp_g, fs_o, vp_o, expect);
  k_cell<<<dim3(BATCH_N / 256), dim3(256), 0, stream>>>(
      expect, c_mag, c_phase, Wo, bo, out);
}
